// TinyLLM_2929167695984
// MI455X (gfx1250) — matrix operand order repaired and checked by reading, not compiled, not run on hardware
//
#include <hip/hip_runtime.h>
#include <hip/hip_bf16.h>
#include <stdint.h>

// ---------------------------------------------------------------------------
// TinyLLM forward pass for MI455X (gfx1250, wave32, WMMA).
// All GEMMs run through v_wmma_f32_16x16x32_f16 (fp16 inputs, fp32 accum).
// - B operands with ragged N (Wp) are padded to a multiple of 64: the GEMM
//   inner loop has zero bounds checks / zero predicated loads.
// - Causal structure is exploited: QK^T skips upper-triangle blocks,
//   P@V clamps the K loop to the causal frontier.
// - Pure-copy LDS staging uses GLOBAL_LOAD_ASYNC_TO_LDS_B128 when the
//   toolchain exposes the builtin (ASYNCcnt-tracked, no VGPR round-trip).
// ---------------------------------------------------------------------------

typedef __attribute__((ext_vector_type(16))) _Float16 v16h;
typedef __attribute__((ext_vector_type(8)))  float    v8f;
typedef __attribute__((ext_vector_type(4)))  int      v4i;

#define TT     2048
#define EMBD   1024
#define VOCABD 50257
#define VPAD   50304   // VOCABD rounded up to multiple of 64
#define NROWS  4096    // B * T

#if defined(__HIP_DEVICE_COMPILE__) && \
    __has_builtin(__builtin_amdgcn_global_load_async_to_lds_b128) && \
    __has_builtin(__builtin_amdgcn_s_wait_asynccnt)
#define USE_ASYNC 1
#else
#define USE_ASYNC 0
#endif

#define GAS __attribute__((address_space(1)))
#define LAS __attribute__((address_space(3)))

// ---------------------------------------------------------------------------
// fp32 -> fp16 convert (square weights)
// ---------------------------------------------------------------------------
__global__ __launch_bounds__(256)
void cvt_f32_f16(const float* __restrict__ src, _Float16* __restrict__ dst, long n) {
  long i = (long)blockIdx.x * blockDim.x + threadIdx.x;
  long stride = (long)gridDim.x * blockDim.x;
  for (; i < n; i += stride) dst[i] = (_Float16)src[i];
}

// ---------------------------------------------------------------------------
// fp32 [rows,N] -> fp16 [rows,Np] with zero padding of columns N..Np
// ---------------------------------------------------------------------------
__global__ __launch_bounds__(256)
void cvt_pad_f32_f16(const float* __restrict__ src, _Float16* __restrict__ dst,
                     int rows, int N, int Np) {
  int c = blockIdx.x * 256 + threadIdx.x;
  int r = blockIdx.y;
  if (c < Np)
    dst[(size_t)r * Np + c] = (c < N) ? (_Float16)src[(size_t)r * N + c]
                                      : (_Float16)0.f;
}

// ---------------------------------------------------------------------------
// x = tok_emb[input_seq] + pos_emb[:T], written as fp16 [4096,1024]
// ---------------------------------------------------------------------------
__global__ __launch_bounds__(256)
void embed_kernel(const int* __restrict__ seq, const float* __restrict__ tok,
                  const float* __restrict__ pos, _Float16* __restrict__ Xh) {
  int row = blockIdx.x;           // 0..4095 = b*T + t
  int t = row & (TT - 1);
  int idx = seq[row];
  const float* tr = tok + (size_t)idx * EMBD;
  const float* pr = pos + (size_t)t   * EMBD;
  _Float16*    xr = Xh  + (size_t)row * EMBD;
  for (int e = threadIdx.x; e < EMBD; e += 256)
    xr[e] = (_Float16)(tr[e] + pr[e]);
}

// ---------------------------------------------------------------------------
// WMMA GEMM: C[M,N] = A[M,K] * B(K,N)  (B read transposed if TRANS_B: B[N,K])
// Block tile 64x64, K-step 64, 256 threads = 8 waves (wave32).
// Wave (wm,wn) computes rows [wm*16,+16) x cols [wn*32,+32) as two 16x16
// accumulators; 4 x v_wmma_f32_16x16x32_f16 per K-stage (2 ksub x 2 nsub).
// LDS: A staged row-major [m][k]; B staged K-major [n][k] so fragments load
// as aligned b32/b128 LDS reads matching CDNA5 16-bit A/B VGPR layouts.
// CAUSAL_SKIP: retire blocks strictly above the diagonal (QK^T scores).
// CAUSAL_K:    clamp K loop to m0+64 (P@V where P is causal-zero).
// Requirements: M % 64 == 0, K % 64 == 0, B buffer padded so that reads up
// to column n0+63 are in-bounds (caller pads ragged N). lda/ldb even.
// ---------------------------------------------------------------------------
#define SA 72
#define SB 72

union FragU { v16h v; uint32_t u[8]; };

__device__ __forceinline__ void stage16B(const _Float16* src, _Float16* dst) {
#if USE_ASYNC
  GAS v4i* g = reinterpret_cast<GAS v4i*>(reinterpret_cast<uintptr_t>(src));
  LAS v4i* l = reinterpret_cast<LAS v4i*>(reinterpret_cast<uintptr_t>(dst));
  __builtin_amdgcn_global_load_async_to_lds_b128(g, l, 0, 0);
#else
  *(uint4*)dst = *(const uint4*)src;
#endif
}

template<bool TRANS_B, bool OUT_F16, bool ADD_BIAS, bool CAUSAL_SKIP, bool CAUSAL_K>
__global__ __launch_bounds__(256)
void gemm_wmma(const _Float16* __restrict__ A, const _Float16* __restrict__ B,
               void* __restrict__ Cv, const float* __restrict__ bias,
               int M, int N, int K, int lda, int ldb, int ldc,
               long long sA, long long sB, long long sC) {
  __shared__ alignas(16) _Float16 As[64 * SA];
  __shared__ alignas(16) _Float16 Bs[64 * SB];

  const int m0 = blockIdx.y * 64;
  const int n0 = blockIdx.x * 64;

  if (CAUSAL_SKIP && n0 > m0 + 63) return;   // never read by causal softmax

  const int tid  = threadIdx.x;
  const int lane = tid & 31;
  const int wid  = tid >> 5;
  const int wm   = wid & 3;        // 0..3  : 16-row strip
  const int wn   = wid >> 2;       // 0..1  : 32-col strip
  const int g    = lane >> 4;      // lane group (0,1)
  const int l15  = lane & 15;

  A += (long long)blockIdx.z * sA;
  B += (long long)blockIdx.z * sB;

  const int Kend = CAUSAL_K ? (m0 + 64 < K ? m0 + 64 : K) : K;

  v8f acc0 = {};
  v8f acc1 = {};

  for (int k0 = 0; k0 < Kend; k0 += 64) {
    // ---- A tile 64x64: 16 halves per thread, async-to-LDS when available
    {
      int row = tid >> 2;             // 0..63
      int col = (tid & 3) * 16;       // 0,16,32,48
      const _Float16* src = A + (size_t)(m0 + row) * lda + (k0 + col);
      _Float16* dst = &As[row * SA + col];
      stage16B(src, dst);
      stage16B(src + 8, dst + 8);
    }
    // ---- B tile into Bs[n][k]
    if (TRANS_B) {
      // B stored [N,K] row-major: straight b128 copies (async when available)
      int nn = tid >> 2;              // 0..63
      int kk = (tid & 3) * 16;        // 0,16,32,48
      const _Float16* src = B + (size_t)(n0 + nn) * ldb + (k0 + kk);
      _Float16* dst = &Bs[nn * SB + kk];
      stage16B(src, dst);
      stage16B(src + 8, dst + 8);
    } else {
      // B stored [K,N] row-major (N pre-padded): transpose-stage with
      // packed (k,k+1) dword pairs -> 2 b128 global loads + ds_store_2addr.
      int kp = tid >> 3;              // 0..31 -> k = 2*kp
      int nn = (tid & 7) * 8;         // 0..56
      const _Float16* r0 = B + (size_t)(k0 + 2 * kp) * ldb + n0 + nn;
      uint4 a0 = *(const uint4*)r0;
      uint4 a1 = *(const uint4*)(r0 + ldb);
      // pull next K-step's rows toward this WGP while WMMAs run (speculative)
      __builtin_prefetch(r0 + (size_t)64 * ldb, 0, 1);
      const uint32_t* pa = (const uint32_t*)&a0;
      const uint32_t* pb = (const uint32_t*)&a1;
      #pragma unroll
      for (int j = 0; j < 4; ++j) {
        uint32_t lo = pa[j], hi = pb[j];
        uint32_t w0 = (lo & 0xffffu) | (hi << 16);
        uint32_t w1 = (lo >> 16) | (hi & 0xffff0000u);
        *(uint32_t*)&Bs[(nn + 2 * j    ) * SB + 2 * kp] = w0;
        *(uint32_t*)&Bs[(nn + 2 * j + 1) * SB + 2 * kp] = w1;
      }
    }
#if USE_ASYNC
    __builtin_amdgcn_s_wait_asynccnt(0);   // LDS writes visible before barrier
#endif
    __syncthreads();

    #pragma unroll
    for (int ks = 0; ks < 64; ks += 32) {
      // A fragment (16x32): lane<16 holds K {0..7,16..23}, lane>=16 +8
      FragU a;
      #pragma unroll
      for (int r = 0; r < 8; ++r) {
        int kb = ((r & 4) << 2) + (r & 3) * 2 + g * 8;
        a.u[r] = *(const uint32_t*)&As[(wm * 16 + l15) * SA + ks + kb];
      }
      // B fragments (32x16): lane<16 holds K {0..7,16..23}, lane>=16 +8
      FragU b0, b1;
      #pragma unroll
      for (int r = 0; r < 8; ++r) {
        int kb = ((r & 4) << 2) + (r & 3) * 2 + g * 8;
        b0.u[r] = *(const uint32_t*)&Bs[(wn * 32 +      l15) * SB + ks + kb];
        b1.u[r] = *(const uint32_t*)&Bs[(wn * 32 + 16 + l15) * SB + ks + kb];
      }
      acc0 = __builtin_amdgcn_wmma_f32_16x16x32_f16(false, a.v, false, b0.v,
                                                    (short)0, acc0, false, false);
      acc1 = __builtin_amdgcn_wmma_f32_16x16x32_f16(false, a.v, false, b1.v,
                                                    (short)0, acc1, false, false);
    }
    __syncthreads();
  }

  // ---- epilogue: C/D layout => lane row m = r + g*8, col n = lane&15
  // Fast path (block fully interior in N) is branch-free: one base pointer,
  // += ldc per row, bias hoisted out of the row loop.
  const long long cb  = (long long)blockIdx.z * sC;
  const int gc0  = n0 + wn * 32 + l15;
  const int gc1  = gc0 + 16;
  const int row0 = m0 + wm * 16 + g * 8;

  float bias0 = 0.f, bias1 = 0.f;
  if (ADD_BIAS) {
    bias0 = (gc0 < N) ? bias[gc0] : 0.f;
    bias1 = (gc1 < N) ? bias[gc1] : 0.f;
  }

  if (n0 + 64 <= N) {              // block-uniform branch
    if (OUT_F16) {
      _Float16* C = (_Float16*)Cv + cb + (size_t)row0 * ldc + gc0;
      #pragma unroll
      for (int r = 0; r < 8; ++r) {
        float v0 = acc0[r]; float v1 = acc1[r];
        if (ADD_BIAS) { v0 += bias0; v1 += bias1; }
        C[0]  = (_Float16)v0;
        C[16] = (_Float16)v1;
        C += ldc;
      }
    } else {
      float* C = (float*)Cv + cb + (size_t)row0 * ldc + gc0;
      #pragma unroll
      for (int r = 0; r < 8; ++r) {
        float v0 = acc0[r]; float v1 = acc1[r];
        if (ADD_BIAS) { v0 += bias0; v1 += bias1; }
        C[0]  = v0;
        C[16] = v1;
        C += ldc;
      }
    }
  } else {                         // ragged-N edge block (logits only)
    #pragma unroll
    for (int r = 0; r < 8; ++r) {
      float v0 = acc0[r]; float v1 = acc1[r];
      if (ADD_BIAS) { v0 += bias0; v1 += bias1; }
      size_t rowoff = (size_t)(row0 + r) * ldc;
      if (OUT_F16) {
        _Float16* C = (_Float16*)Cv + cb;
        if (gc0 < N) C[rowoff + gc0] = (_Float16)v0;
        if (gc1 < N) C[rowoff + gc1] = (_Float16)v1;
      } else {
        float* C = (float*)Cv + cb;
        if (gc0 < N) C[rowoff + gc0] = v0;
        if (gc1 < N) C[rowoff + gc1] = v1;
      }
    }
  }
}

// ---------------------------------------------------------------------------
// Causal softmax over scores row (scale = 1/sqrt(1024) = 1/32), fp16 probs out
// Writes exact zeros for s > t (required by the CAUSAL_K P@V GEMM).
// ---------------------------------------------------------------------------
__global__ __launch_bounds__(256)
void softmax_causal(const float* __restrict__ S, _Float16* __restrict__ P) {
  __shared__ float red[256];
  int rowg = blockIdx.x;          // b*T + t
  int t = rowg & (TT - 1);
  const float* srow = S + (size_t)rowg * TT;
  _Float16*    prow = P + (size_t)rowg * TT;
  const float scale = 0.03125f;

  float m = -3.4e38f;
  for (int s = threadIdx.x; s <= t; s += 256)
    m = fmaxf(m, srow[s] * scale);
  red[threadIdx.x] = m; __syncthreads();
  for (int off = 128; off > 0; off >>= 1) {
    if (threadIdx.x < off)
      red[threadIdx.x] = fmaxf(red[threadIdx.x], red[threadIdx.x + off]);
    __syncthreads();
  }
  m = red[0]; __syncthreads();

  float sum = 0.f;
  for (int s = threadIdx.x; s <= t; s += 256)
    sum += __expf(srow[s] * scale - m);
  red[threadIdx.x] = sum; __syncthreads();
  for (int off = 128; off > 0; off >>= 1) {
    if (threadIdx.x < off)
      red[threadIdx.x] += red[threadIdx.x + off];
    __syncthreads();
  }
  float inv = 1.f / red[0];

  for (int s = threadIdx.x; s < TT; s += 256) {
    float p = (s <= t) ? __expf(srow[s] * scale - m) * inv : 0.f;
    prow[s] = (_Float16)p;
  }
}

// ---------------------------------------------------------------------------
// Cross entropy: mean over rows of (logsumexp(logits) - logits[target])
// ---------------------------------------------------------------------------
__global__ void zero_loss(float* loss) { *loss = 0.f; }

__global__ __launch_bounds__(256)
void ce_kernel(const float* __restrict__ logits, const int* __restrict__ tgt,
               float* __restrict__ loss) {
  __shared__ float red[256];
  int row = blockIdx.x;
  const float* lr = logits + (size_t)row * VOCABD;

  float m = -3.4e38f;
  for (int i = threadIdx.x; i < VOCABD; i += 256) m = fmaxf(m, lr[i]);
  red[threadIdx.x] = m; __syncthreads();
  for (int off = 128; off > 0; off >>= 1) {
    if (threadIdx.x < off)
      red[threadIdx.x] = fmaxf(red[threadIdx.x], red[threadIdx.x + off]);
    __syncthreads();
  }
  m = red[0]; __syncthreads();

  float sum = 0.f;
  for (int i = threadIdx.x; i < VOCABD; i += 256) sum += __expf(lr[i] - m);
  red[threadIdx.x] = sum; __syncthreads();
  for (int off = 128; off > 0; off >>= 1) {
    if (threadIdx.x < off)
      red[threadIdx.x] += red[threadIdx.x + off];
    __syncthreads();
  }
  if (threadIdx.x == 0) {
    float nll = __logf(red[0]) + m - lr[tgt[row]];
    atomicAdd(loss, nll * (1.0f / NROWS));
  }
}

// ---------------------------------------------------------------------------
// Host launcher
// ---------------------------------------------------------------------------
extern "C" void kernel_launch(void* const* d_in, const int* in_sizes, int n_in,
                              void* d_out, int out_size, void* d_ws, size_t ws_size,
                              hipStream_t stream) {
  const int*   seq = (const int*)  d_in[0];
  const int*   tgt = (const int*)  d_in[1];
  const float* tok = (const float*)d_in[2];
  const float* pos = (const float*)d_in[3];
  const float* Wk  = (const float*)d_in[4];
  const float* Wq  = (const float*)d_in[5];
  const float* Wv  = (const float*)d_in[6];
  const float* Wp  = (const float*)d_in[7];
  const float* bp  = (const float*)d_in[8];

  float* logits = (float*)d_out;                       // [4096, 50257]
  float* loss   = logits + (size_t)NROWS * VOCABD;     // 1 element

  // workspace carve-up (256B aligned)
  char* w = (char*)d_ws;
  auto carve = [&](size_t bytes) {
    char* p = w;
    w += (bytes + 255) & ~(size_t)255;
    return p;
  };
  _Float16* Xh  = (_Float16*)carve((size_t)NROWS * EMBD * 2);     // embedded x
  _Float16* Wqh = (_Float16*)carve((size_t)EMBD * EMBD * 2);
  _Float16* Wkh = (_Float16*)carve((size_t)EMBD * EMBD * 2);
  _Float16* Wvh = (_Float16*)carve((size_t)EMBD * EMBD * 2);
  _Float16* Wph = (_Float16*)carve((size_t)EMBD * VPAD * 2);      // padded, L2-resident
  _Float16* Qh  = (_Float16*)carve((size_t)NROWS * EMBD * 2);
  _Float16* Kh  = (_Float16*)carve((size_t)NROWS * EMBD * 2);
  _Float16* Vh  = (_Float16*)carve((size_t)NROWS * EMBD * 2);
  float*    Sc  = (float*)   carve((size_t)2 * TT * TT * 4);      // scores fp32
  _Float16* Ph  = (_Float16*)carve((size_t)2 * TT * TT * 2);      // probs fp16
  _Float16* Ah  = (_Float16*)carve((size_t)NROWS * EMBD * 2);     // attn_out fp16

  // 1) weight converts (+ pad Wp to VPAD columns) + embedding
  cvt_f32_f16<<<1024, 256, 0, stream>>>(Wq, Wqh, (long)EMBD * EMBD);
  cvt_f32_f16<<<1024, 256, 0, stream>>>(Wk, Wkh, (long)EMBD * EMBD);
  cvt_f32_f16<<<1024, 256, 0, stream>>>(Wv, Wvh, (long)EMBD * EMBD);
  {
    dim3 gp((VPAD + 255) / 256, EMBD, 1);
    cvt_pad_f32_f16<<<gp, 256, 0, stream>>>(Wp, Wph, EMBD, VOCABD, VPAD);
  }
  embed_kernel<<<NROWS, 256, 0, stream>>>(seq, tok, pos, Xh);

  // 2) Q/K/V projections: [4096,1024] = Xh @ W[1024,1024]
  dim3 gqkv(EMBD / 64, NROWS / 64, 1);
  gemm_wmma<false, true, false, false, false><<<gqkv, 256, 0, stream>>>(
      Xh, Wqh, Qh, nullptr, NROWS, EMBD, EMBD, EMBD, EMBD, EMBD, 0, 0, 0);
  gemm_wmma<false, true, false, false, false><<<gqkv, 256, 0, stream>>>(
      Xh, Wkh, Kh, nullptr, NROWS, EMBD, EMBD, EMBD, EMBD, EMBD, 0, 0, 0);
  gemm_wmma<false, true, false, false, false><<<gqkv, 256, 0, stream>>>(
      Xh, Wvh, Vh, nullptr, NROWS, EMBD, EMBD, EMBD, EMBD, EMBD, 0, 0, 0);

  // 3) scores = Q @ K^T per batch (fp32 out, upper-triangle blocks skipped),
  //    then causal softmax -> fp16 P
  dim3 gsc(TT / 64, TT / 64, 2);
  gemm_wmma<true, false, false, true, false><<<gsc, 256, 0, stream>>>(
      Qh, Kh, Sc, nullptr, TT, TT, EMBD, EMBD, EMBD, TT,
      (long long)TT * EMBD, (long long)TT * EMBD, (long long)TT * TT);
  softmax_causal<<<2 * TT, 256, 0, stream>>>(Sc, Ph);

  // 4) attn_out = P @ V per batch (fp16 out, K clamped to causal frontier)
  dim3 gav(EMBD / 64, TT / 64, 2);
  gemm_wmma<false, true, false, false, true><<<gav, 256, 0, stream>>>(
      Ph, Vh, Ah, nullptr, TT, EMBD, TT, TT, EMBD, EMBD,
      (long long)TT * TT, (long long)TT * EMBD, (long long)TT * EMBD);

  // 5) logits = attn_out @ Wp(padded) + bp  (fp32 out straight to d_out)
  dim3 glg(VPAD / 64, NROWS / 64, 1);
  gemm_wmma<false, false, true, false, false><<<glg, 256, 0, stream>>>(
      Ah, Wph, logits, bp, NROWS, VOCABD, EMBD, EMBD, VPAD, VOCABD, 0, 0, 0);

  // 6) mean cross-entropy
  zero_loss<<<1, 1, 0, stream>>>(loss);
  ce_kernel<<<NROWS, 256, 0, stream>>>(logits, tgt, loss);
}
